// MSAAttention_63608465654534
// MI455X (gfx1250) — matrix operand order repaired and checked by reading, not compiled, not run on hardware
//
#include <hip/hip_runtime.h>
#include <hip/hip_bf16.h>

typedef __attribute__((ext_vector_type(16))) _Float16 v16h;
typedef __attribute__((ext_vector_type(8)))  _Float16 h8;
typedef __attribute__((ext_vector_type(4)))  _Float16 h4;
typedef __attribute__((ext_vector_type(8)))  float    v8f;

#define WMMA16(a, b, c) \
  __builtin_amdgcn_wmma_f32_16x16x32_f16(false, (a), false, (b), (short)0, (c), false, false)

// ---------- WMMA fragment loaders (gfx1250 wave32 layouts) ----------

// A-matrix 16x32 f16: lane m = lane&15; VGPR0..3 hold K = kh..kh+7,
// VGPR4..7 hold K = 16+kh..16+kh+7, kh = (lane>>4)*8.
static __device__ __forceinline__ v16h frag_a(const _Float16* __restrict__ base,
                                              int ld, int m0, int k0) {
  const int lane = threadIdx.x & 31;
  const int m  = lane & 15;
  const int kh = (lane >> 4) << 3;
  const _Float16* p = base + (size_t)(m0 + m) * (size_t)ld + k0;
  h8 x0 = *(const h8*)(p + kh);
  h8 x1 = *(const h8*)(p + 16 + kh);
  v16h a;
#pragma unroll
  for (int i = 0; i < 8; ++i) { a[i] = x0[i]; a[i + 8] = x1[i]; }
  return a;
}

// B-matrix 32x16 f16, supplied as Bt row-major [N][K] so each column is
// contiguous: lane n = lane&15 holds K = kk..kk+7 and 16+kk..16+kk+7, kk = (lane>>4)*8.
static __device__ __forceinline__ v16h frag_b(const _Float16* __restrict__ base,
                                              int ld, int n0, int k0) {
  const int lane = threadIdx.x & 31;
  const int n  = lane & 15;
  const int kk = (lane >> 4) << 3;
  const _Float16* p = base + (size_t)(n0 + n) * (size_t)ld + k0 + kk;
  h8 x0 = *(const h8*)(p);
  h8 x1 = *(const h8*)(p + 16);
  v16h b;
#pragma unroll
  for (int i = 0; i < 8; ++i) { b[i] = x0[i]; b[i + 8] = x1[i]; }
  return b;
}

// ---------- prep kernels ----------

__global__ void msa_cvt_f16(const float* __restrict__ in, _Float16* __restrict__ out) {
  const int i = blockIdx.x * blockDim.x + threadIdx.x;      // 4 elems/thread
  const float4 v = ((const float4*)in)[i];
  h4 o; o[0] = (_Float16)v.x; o[1] = (_Float16)v.y; o[2] = (_Float16)v.z; o[3] = (_Float16)v.w;
  ((h4*)out)[i] = o;
}

// out[n*K + k] = (f16) in[k*N + n]    (W [K][N] -> Wt [N][K])
__global__ void msa_transpose_f16(const float* __restrict__ in, _Float16* __restrict__ out,
                                  int K, int N) {
  const int i = blockIdx.x * blockDim.x + threadIdx.x;
  const int nn = i / K, kk = i - nn * K;
  out[i] = (_Float16)in[(size_t)kk * N + nn];
}

// ---------- QKV projection: (32768 x 256) @ (256 x 768) + bias ----------

__global__ void msa_qkv_kernel(const _Float16* __restrict__ X,
                               const _Float16* __restrict__ WqT,  // [768][256]
                               const float* __restrict__ bias,
                               _Float16* __restrict__ Q,   // [bh][n][32]
                               _Float16* __restrict__ K,   // [bh][n][32]
                               _Float16* __restrict__ Vt)  // [bh][32][512]
{
  const int wave = threadIdx.x >> 5;
  const int wg   = blockIdx.x * 4 + wave;
  const int nt   = wg % 48, mt = wg / 48;
  const int m0 = mt << 4, n0 = nt << 4;

  v8f acc = {};
#pragma unroll
  for (int kb = 0; kb < 8; ++kb) {
    if (kb < 7) {
      __builtin_prefetch(X + (size_t)m0 * 256 + (kb + 1) * 32, 0, 3);
      __builtin_prefetch(WqT + (size_t)n0 * 256 + (kb + 1) * 32, 0, 3);
    }
    v16h a = frag_a(X, 256, m0, kb * 32);
    v16h b = frag_b(WqT, 256, n0, kb * 32);
    acc = WMMA16(a, b, acc);
  }

  const int lane = threadIdx.x & 31;
  const int n = lane & 15, hi = lane >> 4;
  const int col = n0 + n;
  const float bv = bias[col];
  const int which = col >> 8;      // 0=q 1=k 2=v (uniform within tile)
  const int e = col & 255;
  const int h = e >> 5, d = e & 31;
#pragma unroll
  for (int r = 0; r < 8; ++r) {
    const int row = m0 + r + (hi << 3);        // global token row (0..32767)
    const int b_  = row >> 9, nn = row & 511;
    const int bh  = b_ * 8 + h;
    const _Float16 val = (_Float16)(acc[r] + bv);
    if (which == 0)      Q[((size_t)bh * 512 + nn) * 32 + d] = val;
    else if (which == 1) K[((size_t)bh * 512 + nn) * 32 + d] = val;
    else                 Vt[((size_t)bh * 32 + d) * 512 + nn] = val;
  }
}

// ---------- fused attention: one wave = one (bh, 16-query tile) ----------

__global__ void msa_attn_kernel(const _Float16* __restrict__ Q,
                                const _Float16* __restrict__ K,
                                const _Float16* __restrict__ Vt,
                                const unsigned char* __restrict__ mask,  // [512] bool
                                _Float16* __restrict__ O)                // [32768][256]
{
  __shared__ __align__(16) _Float16 Pst[4][16][32];   // per-wave P staging
  const int wave = threadIdx.x >> 5;
  const int lane = threadIdx.x & 31;
  const int wg = blockIdx.x * 4 + wave;
  const int bh = wg >> 5;
  const int q0 = (wg & 31) << 4;
  const int b_ = bh >> 3, h = bh & 7;
  const int n = lane & 15, hi = lane >> 4;

  const _Float16* Qb = Q  + (size_t)bh * 512 * 32;
  const _Float16* Kb = K  + (size_t)bh * 512 * 32;
  const _Float16* Vb = Vt + (size_t)bh * 32 * 512;

  const v16h aq = frag_a(Qb, 32, q0, 0);          // Q tile, K-dim = d = 32
  v8f o0 = {}, o1 = {};
  float rmax[8], rsum[8];
#pragma unroll
  for (int r = 0; r < 8; ++r) { rmax[r] = -1e30f; rsum[r] = 0.0f; }

  const float SC = 0.17677669529663689f;          // 32^-0.5

  for (int j0 = 0; j0 < 512; j0 += 32) {
    // S = Q * K^T for 32 keys (two 16-col tiles)
    v16h bk0 = frag_b(Kb, 32, j0, 0);
    v16h bk1 = frag_b(Kb, 32, j0 + 16, 0);
    v8f cz = {};
    v8f s0 = WMMA16(aq, bk0, cz);
    v8f s1 = WMMA16(aq, bk1, cz);

    const bool ok0 = mask[j0 + n] != 0;
    const bool ok1 = mask[j0 + 16 + n] != 0;

    float bm[8];
#pragma unroll
    for (int r = 0; r < 8; ++r) {
      float t0 = fminf(fmaxf(s0[r] * SC, -50.0f), 50.0f);
      float t1 = fminf(fmaxf(s1[r] * SC, -50.0f), 50.0f);
      if (!ok0) t0 = -1e30f;
      if (!ok1) t1 = -1e30f;
      s0[r] = t0; s1[r] = t1;
      bm[r] = fmaxf(t0, t1);
    }
    // row-max across the 16 lanes of this half-wave
#pragma unroll
    for (int off = 8; off; off >>= 1)
#pragma unroll
      for (int r = 0; r < 8; ++r)
        bm[r] = fmaxf(bm[r], __shfl_xor(bm[r], off, 32));

    float ps[8];
#pragma unroll
    for (int r = 0; r < 8; ++r) {
      const float nm = fmaxf(rmax[r], bm[r]);
      const float al = __expf(rmax[r] - nm);
      rmax[r] = nm;
      const float p0 = __expf(s0[r] - nm);
      const float p1 = __expf(s1[r] - nm);
      o0[r] *= al; o1[r] *= al;
      rsum[r] *= al;
      ps[r] = p0 + p1;
      const int m = r + (hi << 3);
      Pst[wave][m][n]      = (_Float16)p0;
      Pst[wave][m][n + 16] = (_Float16)p1;
    }
#pragma unroll
    for (int off = 8; off; off >>= 1)
#pragma unroll
      for (int r = 0; r < 8; ++r)
        ps[r] += __shfl_xor(ps[r], off, 32);
#pragma unroll
    for (int r = 0; r < 8; ++r) rsum[r] += ps[r];

    asm volatile("s_wait_dscnt 0" ::: "memory");   // P visible to in-wave reads
    const v16h ap  = frag_a(&Pst[wave][0][0], 32, 0, 0);   // P as A (16x32)
    const v16h bv0 = frag_b(Vb, 512, 0, j0);               // V cols d=0..15
    const v16h bv1 = frag_b(Vb, 512, 16, j0);              // V cols d=16..31
    o0 = WMMA16(ap, bv0, o0);
    o1 = WMMA16(ap, bv1, o1);
    asm volatile("" ::: "memory");
  }

  const size_t orow = (size_t)b_ * 512 + q0;
#pragma unroll
  for (int r = 0; r < 8; ++r) {
    const int m = r + (hi << 3);
    const float inv = 1.0f / rsum[r];
    const size_t base = (orow + m) * 256 + h * 32;
    O[base + n]      = (_Float16)(o0[r] * inv);
    O[base + 16 + n] = (_Float16)(o1[r] * inv);
  }
}

// ---------- output projection: (32768 x 256) @ (256 x 256) + bias ----------

__global__ void msa_oproj_kernel(const _Float16* __restrict__ O,
                                 const _Float16* __restrict__ WoT,  // [256][256]
                                 const float* __restrict__ bo,
                                 float* __restrict__ out)
{
  const int wave = threadIdx.x >> 5;
  const int wg   = blockIdx.x * 4 + wave;
  const int nt = wg & 15, mt = wg >> 4;
  const int m0 = mt << 4, n0 = nt << 4;

  v8f acc = {};
#pragma unroll
  for (int kb = 0; kb < 8; ++kb) {
    if (kb < 7) {
      __builtin_prefetch(O + (size_t)m0 * 256 + (kb + 1) * 32, 0, 3);
      __builtin_prefetch(WoT + (size_t)n0 * 256 + (kb + 1) * 32, 0, 3);
    }
    v16h a = frag_a(O, 256, m0, kb * 32);
    v16h b = frag_b(WoT, 256, n0, kb * 32);
    acc = WMMA16(a, b, acc);
  }

  const int lane = threadIdx.x & 31;
  const int n = lane & 15, hi = lane >> 4;
  const int col = n0 + n;
  const float bv = bo[col];
#pragma unroll
  for (int r = 0; r < 8; ++r) {
    const int row = m0 + r + (hi << 3);
    out[(size_t)row * 256 + col] = acc[r] + bv;
  }
}

// ---------- launch ----------

extern "C" void kernel_launch(void* const* d_in, const int* in_sizes, int n_in,
                              void* d_out, int out_size, void* d_ws, size_t ws_size,
                              hipStream_t stream) {
  const float*         x    = (const float*)d_in[0];
  const unsigned char* mask = (const unsigned char*)d_in[1];
  const float*         Wqkv = (const float*)d_in[2];
  const float*         bqkv = (const float*)d_in[3];
  const float*         Wo   = (const float*)d_in[4];
  const float*         bo   = (const float*)d_in[5];
  float*               out  = (float*)d_out;

  char* ws = (char*)d_ws;
  _Float16* X16  = (_Float16*)(ws);                       // 16 MB (reused as O)
  _Float16* Q16  = (_Float16*)(ws + (size_t)(16u << 20));
  _Float16* K16  = (_Float16*)(ws + (size_t)(32u << 20));
  _Float16* Vt16 = (_Float16*)(ws + (size_t)(48u << 20));
  _Float16* WqT  = (_Float16*)(ws + (size_t)(64u << 20));            // 384 KB
  _Float16* WoT  = (_Float16*)(ws + (size_t)(64u << 20) + (1u << 20)); // 128 KB

  // prep: f16 conversions / weight transposes
  msa_cvt_f16<<<8192, 256, 0, stream>>>(x, X16);                 // 64*512*256
  msa_transpose_f16<<<768, 256, 0, stream>>>(Wqkv, WqT, 256, 768);
  msa_transpose_f16<<<256, 256, 0, stream>>>(Wo, WoT, 256, 256);

  // QKV projection: 2048 M-tiles x 48 N-tiles, 4 waves/block
  msa_qkv_kernel<<<24576, 128, 0, stream>>>(X16, WqT, bqkv, Q16, K16, Vt16);

  // fused attention: 512 bh * 32 q-tiles = 16384 waves (O aliases X16)
  msa_attn_kernel<<<4096, 128, 0, stream>>>(Q16, K16, Vt16, mask, X16);

  // output projection: 2048 x 16 tiles
  msa_oproj_kernel<<<8192, 128, 0, stream>>>(X16, WoT, bo, out);
}
